// AE_spikes_80281528697089
// MI455X (gfx1250) — matrix operand order repaired and checked by reading, not compiled, not run on hardware
//
#include <hip/hip_runtime.h>

typedef __attribute__((ext_vector_type(16))) _Float16      v16h;
typedef __attribute__((ext_vector_type(8)))  float         v8f;
typedef __attribute__((ext_vector_type(8)))  unsigned short v8us;

#define IN_F   784
#define S0K    800      // 784 padded to 25*32
#define HID    128
#define STEPS  16

// ---------------------------------------------------------------------------
// Pack W [N][K] f32 (row-major) into f16 WMMA B-operand blocks.
// Block (ks, nt): per lane l, 16 contiguous halves:
//   n = nt*16 + (l & 15),  k = ks*32 + 8*(l>>4) + j (j < 8), ks*32 + 16 + 8*(l>>4) + (j-8) (j >= 8)
// ---------------------------------------------------------------------------
__global__ void k_pack_b(const float* __restrict__ W, _Float16* __restrict__ dst,
                         int K, int ntiles, int ksteps) {
  int idx   = blockIdx.x * blockDim.x + threadIdx.x;
  int total = ksteps * ntiles * 512;
  if (idx >= total) return;
  int blk = idx >> 9, slot = idx & 511;
  int l = slot >> 4, j = slot & 15;
  int ks = blk / ntiles, nt = blk % ntiles;
  int n = nt * 16 + (l & 15);
  int k = ks * 32 + ((l >> 4) << 3) + (j < 8 ? j : j + 8);
  _Float16 v = (_Float16)0.0f;
  if (k < K) v = (_Float16)W[n * K + k];
  dst[idx] = v;
}

// ---------------------------------------------------------------------------
// Encoder: integrate-and-fire on analog input, all 16 steps bit-packed into
// one u16 per (row, k). k padded to 800 with zeros.
// ---------------------------------------------------------------------------
__global__ void k_encode(const float* __restrict__ feat,
                         unsigned short* __restrict__ s0, int B) {
  int i = blockIdx.x * blockDim.x + threadIdx.x;
  if (i >= B * S0K) return;
  int row = i / S0K, k = i - row * S0K;
  float x = (k < IN_F) ? feat[row * IN_F + k] : 0.0f;
  float v = 0.0f;
  unsigned w = 0;
  #pragma unroll
  for (int t = 0; t < STEPS; ++t) {
    v += x;
    unsigned s = (v >= 1.0f) ? 1u : 0u;
    w |= s << t;
    v -= (float)s;
  }
  s0[i] = (unsigned short)w;
}

// ---------------------------------------------------------------------------
// Phase 1: layers 1-3, full 16-step scan. One wave owns 16 batch rows;
// v1,v2,v3 live in 24 v8f accumulators across the scan. Layer-3 spikes are
// emitted as ballot words: s3raw[wg][t][64] u32.
// ---------------------------------------------------------------------------
__global__ __launch_bounds__(256, 1) void k_layers123(
    const unsigned short* __restrict__ s0,   // [B][800]
    const _Float16* __restrict__ w1p,        // 25*8 blocks
    const _Float16* __restrict__ w2p,        // 4*8 blocks
    const _Float16* __restrict__ w3p,        // 4*8 blocks
    const float* __restrict__ b1, const float* __restrict__ b2,
    const float* __restrict__ b3,
    unsigned int* __restrict__ s3raw) {
  const int lane = threadIdx.x & 31;
  const int wid  = threadIdx.x >> 5;
  const int wg   = blockIdx.x * 8 + wid;   // rows wg*16 .. wg*16+15
  const int m    = lane & 15;
  const int hi   = lane >> 4;
  const int shb  = ((m >> 3) << 4) + (hi << 3);   // bit base for A rebuild

  v8f a1[8], a2[8], a3[8];
  float bv1[8], bv2[8], bv3[8];
  #pragma unroll
  for (int nt = 0; nt < 8; ++nt) {
    v8f z = {0.f,0.f,0.f,0.f,0.f,0.f,0.f,0.f};
    a1[nt] = z; a2[nt] = z; a3[nt] = z;
    bv1[nt] = b1[nt * 16 + m];
    bv2[nt] = b2[nt * 16 + m];
    bv3[nt] = b3[nt * 16 + m];
  }

  const unsigned short* s0row = s0 + (size_t)(wg * 16 + m) * S0K;
  const _Float16* w1l = w1p + lane * 16;
  const _Float16* w2l = w2p + lane * 16;
  const _Float16* w3l = w3p + lane * 16;
  unsigned int* s3w = s3raw + (size_t)wg * (STEPS * 64);

  for (int t = 0; t < STEPS; ++t) {
    // ---------------- layer 1: v1 += s0 @ W1.T ----------------
    #pragma unroll 5
    for (int ks = 0; ks < 25; ++ks) {
      // preload full B set for this k-step (one clause, one wait for 8 WMMAs)
      const _Float16* bb = w1l + (size_t)(ks * 8) * 512;
      v16h Bv[8];
      #pragma unroll
      for (int nt = 0; nt < 8; ++nt) Bv[nt] = *(const v16h*)(bb + nt * 512);
      v8us r0 = *(const v8us*)(s0row + ks * 32 + hi * 8);
      v8us r1 = *(const v8us*)(s0row + ks * 32 + hi * 8 + 16);
      v16h A;
      #pragma unroll
      for (int i = 0; i < 8; ++i) {
        A[i]     = (_Float16)((r0[i] >> t) & 1);
        A[i + 8] = (_Float16)((r1[i] >> t) & 1);
      }
      #pragma unroll
      for (int nt = 0; nt < 8; ++nt)
        a1[nt] = __builtin_amdgcn_wmma_f32_16x16x32_f16(
            false, A, false, Bv[nt], (short)0, a1[nt], false, false);
    }
    // bias + fire + reset; deposit ballot word (nt*8+j) into its lane
    unsigned bw0 = 0, bw1 = 0;
    #pragma unroll
    for (int nt = 0; nt < 8; ++nt) {
      #pragma unroll
      for (int j = 0; j < 8; ++j) {
        float v = a1[nt][j] + bv1[nt];
        bool sp = (v >= 1.0f);
        unsigned mb = __builtin_amdgcn_ballot_w32(sp);
        a1[nt][j] = sp ? (v - 1.0f) : v;
        int widx = nt * 8 + j;
        if (widx < 32) bw0 = (lane == widx)      ? mb : bw0;
        else           bw1 = (lane == widx - 32) ? mb : bw1;
      }
    }

    // ---------------- layer 2: v2 += s1 @ W2.T ----------------
    #pragma unroll
    for (int ks = 0; ks < 4; ++ks) {
      const _Float16* bb = w2l + (size_t)(ks * 8) * 512;
      v16h Bv[8];
      #pragma unroll
      for (int nt = 0; nt < 8; ++nt) Bv[nt] = *(const v16h*)(bb + nt * 512);
      v16h A;
      #pragma unroll
      for (int r = 0; r < 2; ++r) {
        int lidx = ((16 * ks + 8 * r) & 31) + (m & 7);
        int wsrc = __builtin_amdgcn_ds_bpermute(lidx * 4, (int)(ks < 2 ? bw0 : bw1));
        unsigned u = ((unsigned)wsrc) >> shb;
        #pragma unroll
        for (int i = 0; i < 8; ++i) A[r * 8 + i] = (_Float16)((u >> i) & 1);
      }
      #pragma unroll
      for (int nt = 0; nt < 8; ++nt)
        a2[nt] = __builtin_amdgcn_wmma_f32_16x16x32_f16(
            false, A, false, Bv[nt], (short)0, a2[nt], false, false);
    }
    unsigned cw0 = 0, cw1 = 0;
    #pragma unroll
    for (int nt = 0; nt < 8; ++nt) {
      #pragma unroll
      for (int j = 0; j < 8; ++j) {
        float v = a2[nt][j] + bv2[nt];
        bool sp = (v >= 1.0f);
        unsigned mb = __builtin_amdgcn_ballot_w32(sp);
        a2[nt][j] = sp ? (v - 1.0f) : v;
        int widx = nt * 8 + j;
        if (widx < 32) cw0 = (lane == widx)      ? mb : cw0;
        else           cw1 = (lane == widx - 32) ? mb : cw1;
      }
    }

    // ---------------- layer 3: v3 += s2 @ W3.T ----------------
    #pragma unroll
    for (int ks = 0; ks < 4; ++ks) {
      const _Float16* bb = w3l + (size_t)(ks * 8) * 512;
      v16h Bv[8];
      #pragma unroll
      for (int nt = 0; nt < 8; ++nt) Bv[nt] = *(const v16h*)(bb + nt * 512);
      v16h A;
      #pragma unroll
      for (int r = 0; r < 2; ++r) {
        int lidx = ((16 * ks + 8 * r) & 31) + (m & 7);
        int wsrc = __builtin_amdgcn_ds_bpermute(lidx * 4, (int)(ks < 2 ? cw0 : cw1));
        unsigned u = ((unsigned)wsrc) >> shb;
        #pragma unroll
        for (int i = 0; i < 8; ++i) A[r * 8 + i] = (_Float16)((u >> i) & 1);
      }
      #pragma unroll
      for (int nt = 0; nt < 8; ++nt)
        a3[nt] = __builtin_amdgcn_wmma_f32_16x16x32_f16(
            false, A, false, Bv[nt], (short)0, a3[nt], false, false);
    }
    unsigned dw0 = 0, dw1 = 0;
    #pragma unroll
    for (int nt = 0; nt < 8; ++nt) {
      #pragma unroll
      for (int j = 0; j < 8; ++j) {
        float v = a3[nt][j] + bv3[nt];
        bool sp = (v >= 1.0f);
        unsigned mb = __builtin_amdgcn_ballot_w32(sp);
        a3[nt][j] = sp ? (v - 1.0f) : v;
        int widx = nt * 8 + j;
        if (widx < 32) dw0 = (lane == widx)      ? mb : dw0;
        else           dw1 = (lane == widx - 32) ? mb : dw1;
      }
    }
    // emit layer-3 spike ballots for phase 2
    s3w[t * 64 + lane]      = dw0;
    s3w[t * 64 + 32 + lane] = dw1;
  }
}

// ---------------------------------------------------------------------------
// Phase 2: layer 4 (128 -> 784) + spike-count. One wave owns 16 rows x 7
// N-tiles (grid.y covers the 49 tiles); v4 and count live in registers.
// ---------------------------------------------------------------------------
__global__ __launch_bounds__(256, 1) void k_layer4(
    const unsigned int* __restrict__ s3raw,  // [B/16][16][64]
    const _Float16* __restrict__ w4p,        // 4*49 blocks
    const float* __restrict__ b4,
    float* __restrict__ out, int B) {
  const int lane = threadIdx.x & 31;
  const int wid  = threadIdx.x >> 5;
  const int wg   = blockIdx.x * 8 + wid;
  const int ntg  = blockIdx.y;             // 0..6, tiles ntg*7 .. ntg*7+6
  const int m    = lane & 15;
  const int hi   = lane >> 4;
  const int shb  = ((m >> 3) << 4) + (hi << 3);

  v8f vmem[7], cnt[7];
  float bv[7];
  #pragma unroll
  for (int ni = 0; ni < 7; ++ni) {
    v8f z = {0.f,0.f,0.f,0.f,0.f,0.f,0.f,0.f};
    vmem[ni] = z; cnt[ni] = z;
    bv[ni] = b4[(ntg * 7 + ni) * 16 + m];
  }

  const unsigned int* sr = s3raw + (size_t)wg * (STEPS * 64);
  const _Float16* w4l = w4p + lane * 16;

  for (int t = 0; t < STEPS; ++t) {
    // hoist all 8 ballot words for this timestep (one clause)
    unsigned wbits[8];
    #pragma unroll
    for (int q = 0; q < 8; ++q) wbits[q] = sr[t * 64 + 8 * q + (m & 7)];

    #pragma unroll
    for (int ks = 0; ks < 4; ++ks) {
      v16h Bv[7];
      #pragma unroll
      for (int ni = 0; ni < 7; ++ni)
        Bv[ni] = *(const v16h*)(w4l + (size_t)(ks * 49 + ntg * 7 + ni) * 512);
      v16h A;
      #pragma unroll
      for (int r = 0; r < 2; ++r) {
        unsigned u = wbits[2 * ks + r] >> shb;
        #pragma unroll
        for (int i = 0; i < 8; ++i) A[r * 8 + i] = (_Float16)((u >> i) & 1);
      }
      #pragma unroll
      for (int ni = 0; ni < 7; ++ni)
        vmem[ni] = __builtin_amdgcn_wmma_f32_16x16x32_f16(
            false, A, false, Bv[ni], (short)0, vmem[ni], false, false);
    }
    #pragma unroll
    for (int ni = 0; ni < 7; ++ni) {
      #pragma unroll
      for (int j = 0; j < 8; ++j) {
        float v = vmem[ni][j] + bv[ni];
        bool sp = (v >= 1.0f);
        vmem[ni][j] = sp ? (v - 1.0f) : v;
        cnt[ni][j] += sp ? 1.0f : 0.0f;
      }
    }
  }

  // rates: out[row][col], C layout row = j + 8*hi, col = tile*16 + m
  #pragma unroll
  for (int ni = 0; ni < 7; ++ni) {
    int col = (ntg * 7 + ni) * 16 + m;
    #pragma unroll
    for (int j = 0; j < 8; ++j) {
      int row = wg * 16 + j + 8 * hi;
      out[(size_t)row * IN_F + col] = cnt[ni][j] * (1.0f / 16.0f);
    }
  }
}

// ---------------------------------------------------------------------------
extern "C" void kernel_launch(void* const* d_in, const int* in_sizes, int n_in,
                              void* d_out, int out_size, void* d_ws, size_t ws_size,
                              hipStream_t stream) {
  const float* feat = (const float*)d_in[0];
  const float* W1 = (const float*)d_in[1];
  const float* b1 = (const float*)d_in[2];
  const float* W2 = (const float*)d_in[3];
  const float* b2 = (const float*)d_in[4];
  const float* W3 = (const float*)d_in[5];
  const float* b3 = (const float*)d_in[6];
  const float* W4 = (const float*)d_in[7];
  const float* b4 = (const float*)d_in[8];
  float* out = (float*)d_out;

  const int B = in_sizes[0] / IN_F;   // 16384

  // workspace layout (bytes)
  char* ws = (char*)d_ws;
  _Float16* w1p = (_Float16*)(ws);                       // 25*8*512*2 = 204800
  _Float16* w2p = (_Float16*)(ws + 204800);              // 32768
  _Float16* w3p = (_Float16*)(ws + 237568);              // 32768
  _Float16* w4p = (_Float16*)(ws + 270336);              // 4*49*512*2 = 200704
  unsigned short* s0b = (unsigned short*)(ws + 471040);  // B*800*2
  unsigned int* s3r = (unsigned int*)(ws + 471040 + (size_t)B * S0K * 2);

  // weight packing
  k_pack_b<<<(25 * 8 * 512 + 255) / 256, 256, 0, stream>>>(W1, w1p, IN_F, 8, 25);
  k_pack_b<<<(4 * 8 * 512 + 255) / 256, 256, 0, stream>>>(W2, w2p, HID, 8, 4);
  k_pack_b<<<(4 * 8 * 512 + 255) / 256, 256, 0, stream>>>(W3, w3p, HID, 8, 4);
  k_pack_b<<<(4 * 49 * 512 + 255) / 256, 256, 0, stream>>>(W4, w4p, HID, 49, 4);

  // encoder spike trains (bit-packed over all 16 steps)
  k_encode<<<((size_t)B * S0K + 255) / 256, 256, 0, stream>>>(feat, s0b, B);

  // layers 1-3 scan
  k_layers123<<<B / 128, 256, 0, stream>>>(s0b, w1p, w2p, w3p, b1, b2, b3, s3r);

  // layer 4 scan + rate output
  k_layer4<<<dim3(B / 128, 7), 256, 0, stream>>>(s3r, w4p, b4, out, B);
}
